// EncoderLayer_49340584297017
// MI455X (gfx1250) — matrix operand order repaired and checked by reading, not compiled, not run on hardware
//
#include <hip/hip_runtime.h>
#include <math.h>

// ---------------------------------------------------------------------------
// Performer (FAVOR+) encoder layer for MI455X / gfx1250.
// GEMMs on v_wmma_f32_16x16x32_bf16; tile staging via Tensor Data Mover
// (TENSOR_LOAD_TO_LDS + s_wait_tensorcnt) and async global->LDS copies
// (GLOBAL_LOAD_ASYNC_TO_LDS_B128 + s_wait_asynccnt), double-buffered so the
// DMA of tile k+1 overlaps the WMMAs of tile k.
// ---------------------------------------------------------------------------

namespace {

constexpr int Bb  = 4;
constexpr int S   = 8192;
constexpr int D   = 512;
constexpr int H   = 8;
constexpr int DH  = 64;
constexpr int DFF = 2048;
constexpr int M   = 256;
constexpr int N   = Bb * S;     // 32768 tokens
constexpr int BH  = Bb * H;     // 32
constexpr int QKVLD = 3 * D;    // 1536

constexpr float NORM  = 0.35355339059327373f;  // DH^-0.25
constexpr float RATIO = 0.0625f;               // M^-0.5
constexpr float EPSK  = 1e-4f;
constexpr float EPSLN = 1e-5f;

typedef __bf16 bf16;
typedef __attribute__((ext_vector_type(16))) __bf16 bf16x16;
typedef __attribute__((ext_vector_type(8)))  __bf16 bf16x8;
typedef __attribute__((ext_vector_type(8)))  float  f32x8;

__device__ __forceinline__ f32x8 wmma_bf16(bf16x16 a, bf16x16 b, f32x8 c) {
  return __builtin_amdgcn_wmma_f32_16x16x32_bf16(false, a, false, b, (short)0, c,
                                                 false, false);
}

__device__ __forceinline__ bf16x16 join16(bf16x8 lo, bf16x8 hi) {
  bf16x16 v;
#pragma unroll
  for (int e = 0; e < 8; ++e) { v[e] = lo[e]; v[e + 8] = hi[e]; }
  return v;
}

// Order-preserving float <-> uint key (for atomicMax on floats of any sign).
__device__ __forceinline__ unsigned fkey(float f) {
  unsigned u = __float_as_uint(f);
  return (u & 0x80000000u) ? ~u : (u | 0x80000000u);
}
__device__ __forceinline__ float funkey(unsigned k) {
  unsigned u = (k & 0x80000000u) ? (k & 0x7fffffffu) : ~k;
  return __uint_as_float(u);
}

// --------------------- CDNA5 data-mover feature guards ---------------------
#if __has_builtin(__builtin_amdgcn_global_load_async_to_lds_b128)
#define HAVE_ASYNC 1
// Prototype (from clang-22 diagnostic): param0 = v4i in addrspace(1) ("__device__"),
// param1 = v4i in addrspace(3), then imm offset + imm cpol.
typedef int v4i_vs __attribute__((__vector_size__(16)));
typedef __attribute__((address_space(1))) v4i_vs* gv4_ptr;
typedef __attribute__((address_space(3))) v4i_vs* lv4_ptr;
__device__ __forceinline__ void async_cp16(const bf16* g, bf16* l) {
  __builtin_amdgcn_global_load_async_to_lds_b128((gv4_ptr)(void*)g,
                                                 (lv4_ptr)(void*)l, 0, 0);
}
__device__ __forceinline__ void async_wait0() {
#if __has_builtin(__builtin_amdgcn_s_wait_asynccnt)
  __builtin_amdgcn_s_wait_asynccnt(0);
#else
  asm volatile("s_wait_asynccnt 0x0" ::: "memory");
#endif
}
#else
#define HAVE_ASYNC 0
__device__ __forceinline__ void async_wait0() {}
#endif

#if __has_builtin(__builtin_amdgcn_tensor_load_to_lds)
#define HAVE_TDM 1
typedef unsigned int u32x4 __attribute__((ext_vector_type(4)));
typedef int i32x8a __attribute__((ext_vector_type(8)));
typedef int i32x4a __attribute__((ext_vector_type(4)));

typedef const __attribute__((address_space(3))) void* las_cptr;
__device__ __forceinline__ unsigned lds_off(const void* p) {
  return (unsigned)(unsigned long)(las_cptr)p;
}

// 2D TDM tile load: 32x128 bf16 tile, row stride `stride_elems` in memory,
// DMA-applied LDS padding: interval 16 DWORDs (64B row) + 4 DWORDs (16B) pad
// -> LDS row stride of 40 bf16 elements, matching the WMMA fragment reads.
__device__ __forceinline__ void tdm_load_tile(const bf16* gsrc, unsigned lds_byte,
                                              long stride_elems, int nrows) {
  unsigned long ga = (unsigned long)gsrc;
  u32x4 g0;
  g0[0] = 1u;                                             // count=1 (valid D#)
  g0[1] = lds_byte;                                       // lds_addr
  g0[2] = (unsigned)ga;                                   // global_addr[31:0]
  g0[3] = (unsigned)((ga >> 32) & 0x01ffffffu) | (2u << 30);  // addr hi | type=2
  unsigned td0 = (unsigned)stride_elems;                  // tensor_dim0
  unsigned td1 = (unsigned)nrows;                         // tensor_dim1
  i32x8a g1;
  g1[0] = (int)((1u << 16)                 // data_size = 2 bytes
              | (1u << 20)                 // pad_enable
              | (3u << 22)                 // pad_interval: 16 DWORDs
              | (3u << 25));               // pad_amount:   4 DWORDs
  g1[1] = (int)((td0 & 0xffffu) << 16);                       // td0[15:0]
  g1[2] = (int)((td0 >> 16) | ((td1 & 0xffffu) << 16));       // td0 hi | td1 lo
  g1[3] = (int)((td1 >> 16) | (32u << 16));                   // td1 hi | tile_dim0=32
  g1[4] = (int)128u;                                          // tile_dim1=128
  g1[5] = (int)td0;                                           // dim0_stride[31:0]
  g1[6] = 0;
  g1[7] = 0;
  i32x4a z4 = {};
#if __clang_major__ >= 23
  i32x8a z8 = {};
  __builtin_amdgcn_tensor_load_to_lds(g0, g1, z4, z4, z8, 0);
#else
  __builtin_amdgcn_tensor_load_to_lds(g0, g1, z4, z4, 0);
#endif
}
__device__ __forceinline__ void tensor_wait0() {
#if __has_builtin(__builtin_amdgcn_s_wait_tensorcnt)
  __builtin_amdgcn_s_wait_tensorcnt(0);
#else
  asm volatile("s_wait_tensorcnt 0x0" ::: "memory");
#endif
}
#else
#define HAVE_TDM 0
__device__ __forceinline__ void tensor_wait0() {}
#endif

// ---------------------------------------------------------------------------
// fp32 -> bf16 conversion (optionally scaled; folds FAVOR `norm` into proj).
// ---------------------------------------------------------------------------
__global__ __launch_bounds__(256) void cvt_bf16(const float* __restrict__ src,
                                                bf16* __restrict__ dst, long n,
                                                float scale) {
  long i = (long)blockIdx.x * 256 + threadIdx.x;
  long stride = (long)gridDim.x * 256;
  for (; i < n; i += stride) dst[i] = (bf16)(src[i] * scale);
}

// ---------------------------------------------------------------------------
// Generic TN GEMM: C[row, col] = sum_k A[row,k] * W[col,k] + bias[col]
//   block tile 128x128, BK=32, 8 waves (2x4), each wave 64x32 = 4x2 frags.
//   Double-buffered LDS; tiles staged by the Tensor Data Mover when present.
//   mode 0: bf16 out, 1: ReLU+bf16 out, 2: fp32 out.
// ---------------------------------------------------------------------------
__global__ __launch_bounds__(256) void gemm_tn(const bf16* __restrict__ A,
                                               const bf16* __restrict__ W,
                                               const float* __restrict__ bias,
                                               void* __restrict__ Cout, int K,
                                               int ldA, int ldC, int mode) {
  __shared__ __align__(16) bf16 sA[2][128 * 40];  // padded stride 40 (80B)
  __shared__ __align__(16) bf16 sB[2][128 * 40];
  int tid = threadIdx.x, wave = tid >> 5, lane = tid & 31;
  int hf = lane >> 4, ln = lane & 15;
  long m0 = (long)blockIdx.x * 128;
  long n0 = (long)blockIdx.y * 128;
  int wm = (wave & 1) * 64;
  int wn = (wave >> 1) * 32;
  int lr = tid >> 1, lc = (tid & 1) * 16;
  (void)lr; (void)lc;

  f32x8 acc[4][2] = {};

  auto issue = [&](int buf, int k0) {
#if HAVE_TDM
    if (wave == 0) {
      tdm_load_tile(A + m0 * ldA + k0, lds_off(&sA[buf][0]), ldA, 128);
      tdm_load_tile(W + n0 * (long)K + k0, lds_off(&sB[buf][0]), K, 128);
    }
#else
    const bf16* sa = A + (m0 + lr) * ldA + k0 + lc;
    const bf16* sb = W + (n0 + lr) * (long)K + k0 + lc;
    *(bf16x8*)&sA[buf][lr * 40 + lc]     = *(const bf16x8*)sa;
    *(bf16x8*)&sA[buf][lr * 40 + lc + 8] = *(const bf16x8*)(sa + 8);
    *(bf16x8*)&sB[buf][lr * 40 + lc]     = *(const bf16x8*)sb;
    *(bf16x8*)&sB[buf][lr * 40 + lc + 8] = *(const bf16x8*)(sb + 8);
#endif
  };

  issue(0, 0);
  int nk = K >> 5;
  for (int ik = 0; ik < nk; ++ik) {
    int buf = ik & 1;
    tensor_wait0();      // current buffer's DMA complete (nop on fallback)
    __syncthreads();     // ..visible to all waves; prev compute done everywhere
    if (ik + 1 < nk) issue(buf ^ 1, (ik + 1) << 5);  // overlap DMA with WMMA

    bf16x16 af[4], bfr[2];
#pragma unroll
    for (int i = 0; i < 4; ++i) {
      const bf16* p = &sA[buf][(wm + i * 16 + ln) * 40];
      af[i] = join16(*(const bf16x8*)(p + hf * 8),
                     *(const bf16x8*)(p + 16 + hf * 8));
    }
#pragma unroll
    for (int j = 0; j < 2; ++j) {
      const bf16* p = &sB[buf][(wn + j * 16 + ln) * 40];
      bfr[j] = join16(*(const bf16x8*)(p + hf * 8), *(const bf16x8*)(p + 16 + hf * 8));
    }
#pragma unroll
    for (int i = 0; i < 4; ++i)
#pragma unroll
      for (int j = 0; j < 2; ++j) acc[i][j] = wmma_bf16(af[i], bfr[j], acc[i][j]);
  }

#pragma unroll
  for (int i = 0; i < 4; ++i)
#pragma unroll
    for (int j = 0; j < 2; ++j) {
      int col = (int)n0 + wn + j * 16 + ln;
      float bv = bias[col];
#pragma unroll
      for (int v = 0; v < 8; ++v) {
        long row = m0 + wm + i * 16 + v + hf * 8;
        float val = acc[i][j][v] + bv;
        if (mode == 1) val = fmaxf(val, 0.f);
        if (mode == 2) ((float*)Cout)[row * ldC + col] = val;
        else           ((bf16*)Cout)[row * ldC + col] = (bf16)val;
      }
    }
}

// ---------------------------------------------------------------------------
// diag[bh,s] = 0.5 * norm^2 * sum(data^2) for q and k.
// ---------------------------------------------------------------------------
__global__ __launch_bounds__(256) void diag_kernel(const bf16* __restrict__ qkv,
                                                   float* __restrict__ qdiag,
                                                   float* __restrict__ kdiag) {
  long idx = (long)blockIdx.x * 256 + threadIdx.x;  // over N*H
  long n = idx >> 3;
  int h = (int)(idx & 7);
  int b = (int)(n >> 13);
  int s = (int)(n & (S - 1));
  const bf16* q = qkv + n * QKVLD + h * DH;
  const bf16* k = q + D;
  float sq = 0.f, sk = 0.f;
#pragma unroll 8
  for (int i = 0; i < DH; ++i) {
    float a = (float)q[i]; sq += a * a;
    float c = (float)k[i]; sk += c * c;
  }
  long o = ((long)(b * H + h) << 13) | s;
  const float cmul = 0.5f * NORM * NORM;
  qdiag[o] = sq * cmul;
  kdiag[o] = sk * cmul;
}

// ---------------------------------------------------------------------------
// FAVOR+ feature kernel. One workgroup: 64 rows x all M=256 features, K=64.
//   mode 0: query (per-row stabilizer, write qp)
//   mode 1: key pass 1 (global max -> kstab[bh] via uint key atomicMax)
//   mode 2: key pass 2 (use kstab[bh], write kp)
// ---------------------------------------------------------------------------
__global__ __launch_bounds__(256) void feature_kernel(
    const bf16* __restrict__ qkv, int qoff, const bf16* __restrict__ projb,
    const float* __restrict__ diag, unsigned* __restrict__ kstab,
    bf16* __restrict__ out, int mode) {
  __shared__ unsigned rowkey[64];
  int bh = blockIdx.y;
  int b = bh >> 3, h = bh & 7;
  int s0 = blockIdx.x * 64;
  int tid = threadIdx.x, wave = tid >> 5, lane = tid & 31;
  int hf = lane >> 4, ln = lane & 15;
  int wr = (wave & 3) * 16;
  int wc = (wave >> 2) * 128;
  if (tid < 64) rowkey[tid] = 0;
  __syncthreads();

  bf16x16 af[2];
  const bf16* arow =
      qkv + (long)(b * S + s0 + wr + ln) * QKVLD + h * DH + qoff;
#pragma unroll
  for (int ks = 0; ks < 2; ++ks) {
    const bf16* p = arow + ks * 32;
    af[ks] = join16(*(const bf16x8*)(p + hf * 8),
                    *(const bf16x8*)(p + 16 + hf * 8));
  }

  f32x8 acc[8] = {};
#pragma unroll
  for (int j = 0; j < 8; ++j) {
    int colm = wc + j * 16 + ln;
#pragma unroll
    for (int ks = 0; ks < 2; ++ks) {
      const bf16* p = projb + (long)colm * DH + ks * 32;
      bf16x16 bfr = join16(*(const bf16x8*)(p + hf * 8), *(const bf16x8*)(p + 16 + hf * 8));
      acc[j] = wmma_bf16(af[ks], bfr, acc[j]);
    }
  }

  if (mode == 1) {
    float wmax = -3.0e38f;
#pragma unroll
    for (int j = 0; j < 8; ++j)
#pragma unroll
      for (int v = 0; v < 8; ++v) wmax = fmaxf(wmax, acc[j][v]);
#pragma unroll
    for (int mm = 16; mm >= 1; mm >>= 1)
      wmax = fmaxf(wmax, __shfl_xor(wmax, mm, 32));
    if (lane == 0) atomicMax(kstab + bh, fkey(wmax));
    return;
  }

  float stab_s = 0.f;
  if (mode == 2) {
    stab_s = funkey(kstab[bh]);
  } else {
#pragma unroll
    for (int v = 0; v < 8; ++v) {
      float rmax = -3.0e38f;
#pragma unroll
      for (int j = 0; j < 8; ++j) rmax = fmaxf(rmax, acc[j][v]);
#pragma unroll
      for (int mm = 8; mm >= 1; mm >>= 1)
        rmax = fmaxf(rmax, __shfl_xor(rmax, mm, 32));
      if (ln == 0) atomicMax(&rowkey[wr + v + hf * 8], fkey(rmax));
    }
    __syncthreads();
  }

#pragma unroll
  for (int v = 0; v < 8; ++v) {
    int rloc = wr + v + hf * 8;
    float stab = (mode == 0) ? funkey(rowkey[rloc]) : stab_s;
    float dg = diag[((long)bh << 13) + s0 + rloc];
#pragma unroll
    for (int j = 0; j < 8; ++j) {
      int col = wc + j * 16 + ln;
      float val = RATIO * (__expf(acc[j][v] - dg - stab) + EPSK);
      out[((long)bh * S + s0 + rloc) * M + col] = (bf16)val;
    }
  }
}

// ---------------------------------------------------------------------------
// ctx[bh,m,dh] = sum_s kp[bh,s,m] * v[b,s,h,dh]; also ksum[bh,m] = sum_s kp.
// grid (8 s-chunks, BH); async global->LDS double-buffered tiles; fp32 atomic
// accumulation into the 256x64 ctx.
// ---------------------------------------------------------------------------
__global__ __launch_bounds__(256) void ctx_kernel(const bf16* __restrict__ kp,
                                                  const bf16* __restrict__ qkv,
                                                  float* __restrict__ ctx,
                                                  float* __restrict__ ksum) {
  __shared__ __align__(16) bf16 skp[2][32 * 264];  // 32 s-rows x 256 m (pad 8)
  __shared__ __align__(16) bf16 sv[2][32 * 72];    // 32 s-rows x 64 dh (pad 8)
  int bh = blockIdx.y;
  int b = bh >> 3, h = bh & 7;
  int schunk = blockIdx.x;  // 1024 rows each
  int tid = threadIdx.x, wave = tid >> 5, lane = tid & 31;
  int hf = lane >> 4, ln = lane & 15;
  int wm = wave * 32;  // 8 waves cover m = 0..255
  f32x8 acc[2][4] = {};
  float ks_part = 0.f;
  int lr = tid >> 3;
  int lc = (tid & 7) * 32;
  int vc = (tid & 7) * 8;

  auto issue = [&](int buf, int sg) {
    const bf16* src = kp + ((long)bh * S + sg + lr) * M + lc;
    const bf16* vs =
        qkv + (long)(b * S + sg + lr) * QKVLD + 2 * D + h * DH + vc;
#if HAVE_ASYNC
#pragma unroll
    for (int u = 0; u < 4; ++u)
      async_cp16(src + u * 8, &skp[buf][lr * 264 + lc + u * 8]);
    async_cp16(vs, &sv[buf][lr * 72 + vc]);
#else
#pragma unroll
    for (int u = 0; u < 4; ++u)
      *(bf16x8*)&skp[buf][lr * 264 + lc + u * 8] = *(const bf16x8*)(src + u * 8);
    *(bf16x8*)&sv[buf][lr * 72 + vc] = *(const bf16x8*)vs;
#endif
  };

  issue(0, schunk * 1024);
  for (int it = 0; it < 32; ++it) {
    int buf = it & 1;
    async_wait0();
    __syncthreads();
    if (it + 1 < 32) issue(buf ^ 1, schunk * 1024 + (it + 1) * 32);

#pragma unroll 8
    for (int r = 0; r < 32; ++r) ks_part += (float)skp[buf][r * 264 + tid];

#pragma unroll
    for (int fm = 0; fm < 2; ++fm) {
      int m = wm + fm * 16 + ln;
      bf16x16 a;  // A[m, s] = kp[s, m] : transpose gather from LDS
#pragma unroll
      for (int e = 0; e < 8; ++e) {
        a[e]     = skp[buf][(hf * 8 + e) * 264 + m];
        a[e + 8] = skp[buf][(16 + hf * 8 + e) * 264 + m];
      }
#pragma unroll
      for (int fj = 0; fj < 4; ++fj) {
        int col = fj * 16 + ln;
        bf16x16 bb;  // B[s, dh] = v, column gather
#pragma unroll
        for (int e = 0; e < 16; ++e) bb[e] = sv[buf][((e < 8 ? 0 : 8) + hf * 8 + e) * 72 + col];
        acc[fm][fj] = wmma_bf16(a, bb, acc[fm][fj]);
      }
    }
  }

  atomicAdd(&ksum[bh * M + tid], ks_part);
#pragma unroll
  for (int fm = 0; fm < 2; ++fm)
#pragma unroll
    for (int fj = 0; fj < 4; ++fj)
#pragma unroll
      for (int v = 0; v < 8; ++v) {
        int m = wm + fm * 16 + v + hf * 8;
        int col = fj * 16 + ln;
        atomicAdd(&ctx[((long)bh * M + m) * DH + col], acc[fm][fj][v]);
      }
}

// ---------------------------------------------------------------------------
// d_inv[bh,s] = 1 / (qp[bh,s,:] . ksum[bh,:])
// ---------------------------------------------------------------------------
__global__ __launch_bounds__(256) void dinv_kernel(const bf16* __restrict__ qp,
                                                   const float* __restrict__ ksum,
                                                   float* __restrict__ dinv) {
  long idx = (long)blockIdx.x * 256 + threadIdx.x;  // over BH*S
  int bh = (int)(idx >> 13);
  const bf16* row = qp + idx * M;
  const float* ks = ksum + bh * M;
  float d = 0.f;
#pragma unroll 4
  for (int c = 0; c < M / 8; ++c) {
    bf16x8 q8 = *(const bf16x8*)(row + c * 8);
#pragma unroll
    for (int e = 0; e < 8; ++e) d += (float)q8[e] * ks[c * 8 + e];
  }
  dinv[idx] = 1.0f / d;
}

// ---------------------------------------------------------------------------
// attn[b,s,h,dh] = (qp[bh,s,:] @ ctx[bh,:,dh]) * dinv[bh,s]
// ---------------------------------------------------------------------------
__global__ __launch_bounds__(256) void attn_kernel(const bf16* __restrict__ qp,
                                                   const float* __restrict__ ctx,
                                                   const float* __restrict__ dinv,
                                                   bf16* __restrict__ attn) {
  __shared__ __align__(16) bf16 sctx[256 * 72];  // M x DH (pad 8), bf16
  int bh = blockIdx.y;
  int b = bh >> 3, h = bh & 7;
  int s0 = blockIdx.x * 128;
  int tid = threadIdx.x, wave = tid >> 5, lane = tid & 31;
  int hf = lane >> 4, ln = lane & 15;
  int wr = (wave & 3) * 32;
  int wc = (wave >> 2) * 32;
  {
    const float* src = ctx + (long)bh * M * DH + (long)tid * DH;
#pragma unroll 8
    for (int c = 0; c < DH; ++c) sctx[tid * 72 + c] = (bf16)src[c];
  }
  __syncthreads();

  f32x8 acc[2][2] = {};
  for (int ksi = 0; ksi < 8; ++ksi) {
    int k0 = ksi * 32;
#pragma unroll
    for (int fm = 0; fm < 2; ++fm) {
      const bf16* p = qp + ((long)bh * S + s0 + wr + fm * 16 + ln) * M + k0;
      bf16x16 a = join16(*(const bf16x8*)(p + hf * 8),
                         *(const bf16x8*)(p + 16 + hf * 8));
#pragma unroll
      for (int fj = 0; fj < 2; ++fj) {
        int col = wc + fj * 16 + ln;
        bf16x16 bb;
#pragma unroll
        for (int e = 0; e < 16; ++e) bb[e] = sctx[(k0 + (e < 8 ? 0 : 8) + hf * 8 + e) * 72 + col];
        acc[fm][fj] = wmma_bf16(a, bb, acc[fm][fj]);
      }
    }
  }

#pragma unroll
  for (int fm = 0; fm < 2; ++fm)
#pragma unroll
    for (int v = 0; v < 8; ++v) {
      int srow = s0 + wr + fm * 16 + v + hf * 8;
      float di = dinv[((long)bh << 13) + srow];
      long n = (long)b * S + srow;
#pragma unroll
      for (int fj = 0; fj < 2; ++fj) {
        int col = wc + fj * 16 + ln;
        attn[n * D + h * DH + col] = (bf16)(acc[fm][fj][v] * di);
      }
    }
}

// ---------------------------------------------------------------------------
// LayerNorm over D=512, one wave per token. mode 0: bf16 out, 1: fp32 out.
// ---------------------------------------------------------------------------
__global__ __launch_bounds__(256) void ln_kernel(const float* __restrict__ src,
                                                 const float* __restrict__ g,
                                                 const float* __restrict__ be,
                                                 bf16* outb, float* outf,
                                                 int mode) {
  int tid = threadIdx.x;
  int wave = tid >> 5, ln = tid & 31;
  long t = (long)blockIdx.x * 8 + wave;
  const float* row = src + t * D;
  float x[16];
  float s = 0.f, sq = 0.f;
#pragma unroll
  for (int i = 0; i < 16; ++i) {
    x[i] = row[ln + i * 32];
    s += x[i];
    sq += x[i] * x[i];
  }
#pragma unroll
  for (int mm = 16; mm >= 1; mm >>= 1) {
    s += __shfl_xor(s, mm, 32);
    sq += __shfl_xor(sq, mm, 32);
  }
  float mean = s * (1.0f / D);
  float var = sq * (1.0f / D) - mean * mean;
  float rs = rsqrtf(var + EPSLN);
#pragma unroll
  for (int i = 0; i < 16; ++i) {
    int c = ln + i * 32;
    float v = (x[i] - mean) * rs * g[c] + be[c];
    if (mode == 0) outb[t * D + c] = (bf16)v;
    else           outf[t * D + c] = v;
  }
}

}  // namespace

// ---------------------------------------------------------------------------
extern "C" void kernel_launch(void* const* d_in, const int* in_sizes, int n_in,
                              void* d_out, int out_size, void* d_ws,
                              size_t ws_size, hipStream_t stream) {
  (void)in_sizes; (void)n_in; (void)out_size; (void)ws_size;
  const float* x    = (const float*)d_in[0];
  const float* Wq   = (const float*)d_in[1];
  const float* bq   = (const float*)d_in[2];
  const float* Wk   = (const float*)d_in[3];
  const float* bk   = (const float*)d_in[4];
  const float* Wv   = (const float*)d_in[5];
  const float* bv   = (const float*)d_in[6];
  const float* Wo   = (const float*)d_in[7];
  const float* bo   = (const float*)d_in[8];
  const float* W1   = (const float*)d_in[9];
  const float* b1   = (const float*)d_in[10];
  const float* W2   = (const float*)d_in[11];
  const float* b2   = (const float*)d_in[12];
  const float* g1   = (const float*)d_in[13];
  const float* be1  = (const float*)d_in[14];
  const float* g2   = (const float*)d_in[15];
  const float* be2  = (const float*)d_in[16];
  const float* proj = (const float*)d_in[17];
  float* out = (float*)d_out;

  char* ws = (char*)d_ws;
  size_t cur = 0;
  auto alloc = [&](size_t bytes) -> void* {
    void* p = ws + cur;
    cur = (cur + bytes + 255) & ~(size_t)255;
    return p;
  };

  bf16*     xb    = (bf16*)alloc((size_t)N * D * 2);
  bf16*     qkvb  = (bf16*)alloc((size_t)N * QKVLD * 2);
  bf16*     wqkvb = (bf16*)alloc((size_t)QKVLD * D * 2);
  bf16*     wob   = (bf16*)alloc((size_t)D * D * 2);
  bf16*     w1b   = (bf16*)alloc((size_t)DFF * D * 2);
  bf16*     w2b   = (bf16*)alloc((size_t)D * DFF * 2);
  bf16*     projb = (bf16*)alloc((size_t)M * DH * 2);
  float*    bqkv  = (float*)alloc((size_t)QKVLD * 4);
  float*    qdiag = (float*)alloc((size_t)BH * S * 4);
  float*    kdiag = (float*)alloc((size_t)BH * S * 4);
  bf16*     qp    = (bf16*)alloc((size_t)BH * S * M * 2);
  bf16*     kp    = (bf16*)alloc((size_t)BH * S * M * 2);
  unsigned* kstab = (unsigned*)alloc((size_t)BH * 4);
  float*    ctxb  = (float*)alloc((size_t)BH * M * DH * 4);
  float*    ksum  = (float*)alloc((size_t)BH * M * 4);
  float*    dinv  = (float*)alloc((size_t)BH * S * 4);
  bf16*     attnb = (bf16*)alloc((size_t)N * D * 2);
  bf16*     hb    = (bf16*)alloc((size_t)N * D * 2);
  bf16*     f1b   = (bf16*)alloc((size_t)N * DFF * 2);
  float*    tmp32 = (float*)alloc((size_t)N * D * 4);

  auto cvt = [&](const float* s, bf16* dp, long n, float sc) {
    cvt_bf16<<<dim3((unsigned)((n + 255) / 256)), dim3(256), 0, stream>>>(s, dp, n, sc);
  };

  // ---- precision conversion (fold FAVOR norm into proj) -------------------
  cvt(x, xb, (long)N * D, 1.0f);
  cvt(Wq, wqkvb,                   (long)D * D, 1.0f);
  cvt(Wk, wqkvb + (long)D * D,     (long)D * D, 1.0f);
  cvt(Wv, wqkvb + (long)2 * D * D, (long)D * D, 1.0f);
  cvt(Wo, wob, (long)D * D, 1.0f);
  cvt(W1, w1b, (long)DFF * D, 1.0f);
  cvt(W2, w2b, (long)D * DFF, 1.0f);
  cvt(proj, projb, (long)M * DH, NORM);

  (void)hipMemcpyAsync(bqkv,         bq, D * 4, hipMemcpyDeviceToDevice, stream);
  (void)hipMemcpyAsync(bqkv + D,     bk, D * 4, hipMemcpyDeviceToDevice, stream);
  (void)hipMemcpyAsync(bqkv + 2 * D, bv, D * 4, hipMemcpyDeviceToDevice, stream);
  (void)hipMemsetAsync(kstab, 0, (size_t)BH * 4, stream);
  (void)hipMemsetAsync(ctxb, 0, (size_t)BH * M * DH * 4, stream);
  (void)hipMemsetAsync(ksum, 0, (size_t)BH * M * 4, stream);

  // ---- fused QKV projection (32768 x 512 x 1536) --------------------------
  gemm_tn<<<dim3(N / 128, QKVLD / 128), dim3(256), 0, stream>>>(
      xb, wqkvb, bqkv, qkvb, D, D, QKVLD, 0);

  // ---- Performer features -------------------------------------------------
  diag_kernel<<<dim3(N * H / 256), dim3(256), 0, stream>>>(qkvb, qdiag, kdiag);
  feature_kernel<<<dim3(S / 64, BH), dim3(256), 0, stream>>>(
      qkvb, 0, projb, qdiag, kstab, qp, 0);                 // qp
  feature_kernel<<<dim3(S / 64, BH), dim3(256), 0, stream>>>(
      qkvb, D, projb, kdiag, kstab, kp, 1);                 // kstab pass
  feature_kernel<<<dim3(S / 64, BH), dim3(256), 0, stream>>>(
      qkvb, D, projb, kdiag, kstab, kp, 2);                 // kp

  // ---- linear attention ---------------------------------------------------
  ctx_kernel<<<dim3(8, BH), dim3(256), 0, stream>>>(kp, qkvb, ctxb, ksum);
  dinv_kernel<<<dim3(BH * S / 256), dim3(256), 0, stream>>>(qp, ksum, dinv);
  attn_kernel<<<dim3(S / 128, BH), dim3(256), 0, stream>>>(qp, ctxb, dinv, attnb);

  // ---- output projection + LN1 --------------------------------------------
  gemm_tn<<<dim3(N / 128, D / 128), dim3(256), 0, stream>>>(
      attnb, wob, bo, tmp32, D, D, D, 2);
  ln_kernel<<<dim3(N / 8), dim3(256), 0, stream>>>(tmp32, g1, be1, hb, nullptr, 0);

  // ---- FFN + LN2 ----------------------------------------------------------
  gemm_tn<<<dim3(N / 128, DFF / 128), dim3(256), 0, stream>>>(
      hb, w1b, b1, f1b, D, D, DFF, 1);
  gemm_tn<<<dim3(N / 128, D / 128), dim3(256), 0, stream>>>(
      f1b, w2b, b2, tmp32, DFF, DFF, D, 2);
  ln_kernel<<<dim3(N / 8), dim3(256), 0, stream>>>(tmp32, g2, be2, nullptr, out, 1);
}
